// GlueVAE_62096637165721
// MI455X (gfx1250) — matrix operand order repaired and checked by reading, not compiled, not run on hardware
//
#include <hip/hip_runtime.h>
#include <math.h>

#define NN 20000
#define EE 160000
#define RR 2000
#define HH 128
#define LL 32
#define EDD 19

typedef __attribute__((ext_vector_type(16))) __bf16 v16bf;
typedef __attribute__((ext_vector_type(8)))  float  v8f;

__device__ __forceinline__ float silu_f(float x) { return x / (1.f + __expf(-x)); }

// ---------------------------------------------------------------------------
// Weight pre-pack: W[K x M] f32 row-major -> bf16 in WMMA B-fragment order.
//   Wp[((kt*M + c)*2 + half)*16 + j] = bf16( W[(kt*32 + (j&8)*2 + half*8 + (j&7))*M + c] )
// so a lane's whole B fragment is 16 contiguous bf16 (32 bytes).
// One block (32 thr) per (kt, c); thread t = half*16 + j. Stores coalesce.
// ---------------------------------------------------------------------------
__global__ void k_pack_w(const float* __restrict__ W, __bf16* __restrict__ Wp, int M)
{
    const int b = blockIdx.x;          // b = kt*M + c
    const int t = threadIdx.x;         // 0..31
    const int kt = b / M, c = b % M;
    Wp[(size_t)b * 32 + t] = (__bf16)W[(size_t)(kt * 32 + ((t & 8) * 2 + (t >> 4) * 8 + (t & 7))) * M + c];
}

// ---------------------------------------------------------------------------
// WMMA GEMM: C[n x M] = act(A[n x K] @ W[K x M] + bias), W pre-packed bf16.
// One 32-lane wave computes a 16 x (TN*16) strip; K compile-time (mult of 32)
// so the K-loop fully unrolls into v_wmma_f32_16x16x32_bf16 chains.
// A fragment: two contiguous 8-float runs per lane -> 4x global_load_b128 +
// v_cvt_pk_bf16_f32. B fragment: one 32-byte vector load from packed weights.
// C/D layout: VGPR r -> row half*8+r, col = lane&15.
// ---------------------------------------------------------------------------
template <int K, int TN, int ACT>
__global__ void k_gemm_t(const float* __restrict__ A, const __bf16* __restrict__ Wp,
                         const float* __restrict__ bias, float* __restrict__ C, int M)
{
    const int lane = threadIdx.x & 31;
    const int half = lane >> 4;
    const int l16  = lane & 15;
    const int rowA = blockIdx.x * 16 + l16;
    const int col0 = blockIdx.y * (TN * 16);

    v8f acc[TN];
#pragma unroll
    for (int t = 0; t < TN; ++t)
        acc[t] = (v8f){0.f, 0.f, 0.f, 0.f, 0.f, 0.f, 0.f, 0.f};

    const float* Arow = A + (size_t)rowA * K;

#pragma unroll
    for (int k0 = 0; k0 < K; k0 += 32) {
        const int kt = k0 / 32;
        // ---- A fragment ----
        const float4 a0 = *(const float4*)(Arow + k0 + half * 8);
        const float4 a1 = *(const float4*)(Arow + k0 + half * 8 + 4);
        const float4 a2 = *(const float4*)(Arow + k0 + 16 + half * 8);
        const float4 a3 = *(const float4*)(Arow + k0 + 16 + half * 8 + 4);
        v16bf a;
        a[0]  = (__bf16)a0.x; a[1]  = (__bf16)a0.y; a[2]  = (__bf16)a0.z; a[3]  = (__bf16)a0.w;
        a[4]  = (__bf16)a1.x; a[5]  = (__bf16)a1.y; a[6]  = (__bf16)a1.z; a[7]  = (__bf16)a1.w;
        a[8]  = (__bf16)a2.x; a[9]  = (__bf16)a2.y; a[10] = (__bf16)a2.z; a[11] = (__bf16)a2.w;
        a[12] = (__bf16)a3.x; a[13] = (__bf16)a3.y; a[14] = (__bf16)a3.z; a[15] = (__bf16)a3.w;

        // ---- B fragments: one 32B vector load per tile ----
#pragma unroll
        for (int t = 0; t < TN; ++t) {
            const v16bf b = *(const v16bf*)(
                Wp + (((size_t)kt * M + col0 + t * 16 + l16) * 2 + half) * 16);
            acc[t] = __builtin_amdgcn_wmma_f32_16x16x32_bf16(
                false, a, false, b, (short)0, acc[t], false, false);
        }
    }

#pragma unroll
    for (int t = 0; t < TN; ++t) {
        const int col = col0 + t * 16 + l16;
        const float bval = bias ? bias[col] : 0.f;
#pragma unroll
        for (int r = 0; r < 8; ++r) {
            const int row = blockIdx.x * 16 + half * 8 + r;
            float val = acc[t][r] + bval;
            if (ACT) val = silu_f(val);
            C[(size_t)row * M + col] = val;
        }
    }
}

template <int K, int TN, int ACT>
static void launch_gemm(const float* A, const __bf16* Wp, const float* b, float* C,
                        int n, int M, hipStream_t stream)
{
    dim3 grid(n / 16, M / (TN * 16));
    k_gemm_t<K, TN, ACT><<<grid, dim3(32), 0, stream>>>(A, Wp, b, C, M);
}

// ---------------------------------------------------------------------------
// Init kernels
// ---------------------------------------------------------------------------
__global__ void k_init_enc(const int* __restrict__ z, const float* __restrict__ embed,
                           const float* __restrict__ vecf,
                           float* __restrict__ s, float* __restrict__ v)
{
    int i = blockIdx.x, t = threadIdx.x;
    s[(size_t)i * HH + t] = embed[(size_t)z[i] * HH + t];
#pragma unroll
    for (int d = 0; d < 3; ++d)
        v[((size_t)i * 3 + d) * HH + t] = vecf[i * 3 + d];
}

__global__ void k_init_dec(const int* __restrict__ z, const float* __restrict__ embed,
                           const int* __restrict__ ridx, const float* __restrict__ rf,
                           float* __restrict__ s)
{
    int i = blockIdx.x, t = threadIdx.x;
    s[(size_t)i * HH + t] = embed[(size_t)z[i] * HH + t] + rf[(size_t)ridx[i] * HH + t];
}

// ---------------------------------------------------------------------------
// Edge geometry (encoder): u = (pos[row]-pos[col]) / dist
// ---------------------------------------------------------------------------
__global__ void k_edge_geom(const int* __restrict__ ei, const float* __restrict__ pos,
                            float* __restrict__ uvec, int E_)
{
    int e = blockIdx.x * blockDim.x + threadIdx.x;
    if (e >= E_) return;
    int r = ei[e], c = ei[E_ + e];
    float dx = pos[r * 3 + 0] - pos[c * 3 + 0];
    float dy = pos[r * 3 + 1] - pos[c * 3 + 1];
    float dz = pos[r * 3 + 2] - pos[c * 3 + 2];
    float dist = sqrtf(dx * dx + dy * dy + dz * dz + 1e-8f);
    uvec[e * 3 + 0] = dx / dist;
    uvec[e * 3 + 1] = dy / dist;
    uvec[e * 3 + 2] = dz / dist;
}

__global__ void k_fake_pos(const float* __restrict__ pos, const float* __restrict__ noise,
                           float* __restrict__ fpos, int n3)
{
    int i = blockIdx.x * blockDim.x + threadIdx.x;
    if (i < n3) fpos[i] = pos[i] + 4.f * noise[i];
}

// Decoder edges: u from fake_pos + fake_edge_attr = [edge_type(3), rbf(16)]
__global__ void k_fake_edge(const int* __restrict__ ei, const float* __restrict__ fpos,
                            const float* __restrict__ eattr, float* __restrict__ fea,
                            float* __restrict__ uvec, int E_)
{
    int e = blockIdx.x * blockDim.x + threadIdx.x;
    if (e >= E_) return;
    int r = ei[e], c = ei[E_ + e];
    float dx = fpos[r * 3 + 0] - fpos[c * 3 + 0];
    float dy = fpos[r * 3 + 1] - fpos[c * 3 + 1];
    float dz = fpos[r * 3 + 2] - fpos[c * 3 + 2];
    float dist = sqrtf(dx * dx + dy * dy + dz * dz + 1e-8f);
    uvec[e * 3 + 0] = dx / dist;
    uvec[e * 3 + 1] = dy / dist;
    uvec[e * 3 + 2] = dz / dist;
    size_t base = (size_t)e * EDD;
    fea[base + 0] = eattr[base + 0];
    fea[base + 1] = eattr[base + 1];
    fea[base + 2] = eattr[base + 2];
    const float coeff = -1.125f;            // -0.5 / (10/15)^2
    const float step = 10.f / 15.f;         // linspace(0,10,16)
#pragma unroll
    for (int j = 0; j < 16; ++j) {
        float d = dist - step * (float)j;
        fea[base + 3 + j] = __expf(coeff * d * d);
    }
}

// ---------------------------------------------------------------------------
// Fused edge message: filt = eattr @ Wf + bf (K=19), x = phi[col]*filt,
// scatter-add ds into s, dv into v. One block (128 thr) per edge, thread t = h.
// ---------------------------------------------------------------------------
__global__ void k_edge_msg(const int* __restrict__ ei, const float* __restrict__ eattr,
                           const float* __restrict__ Wf, const float* __restrict__ bf,
                           const float* __restrict__ phi, const float* __restrict__ uvec,
                           const float* __restrict__ v_old,
                           float* __restrict__ s, float* __restrict__ v, int E_)
{
    __shared__ float ea[32];
    int e = blockIdx.x, t = threadIdx.x;
    if (t < EDD) ea[t] = eattr[(size_t)e * EDD + t];
    __syncthreads();
    int r = ei[e], c = ei[E_ + e];

    float f0 = bf[t], f1 = bf[HH + t], f2 = bf[2 * HH + t];
#pragma unroll
    for (int k = 0; k < EDD; ++k) {
        float ek = ea[k];
        f0 = fmaf(ek, Wf[k * 3 * HH + t],          f0);
        f1 = fmaf(ek, Wf[k * 3 * HH + HH + t],     f1);
        f2 = fmaf(ek, Wf[k * 3 * HH + 2 * HH + t], f2);
    }
    size_t pc = (size_t)c * 3 * HH;
    float ds = phi[pc + t] * f0;
    float x1 = phi[pc + HH + t] * f1;
    float x2 = phi[pc + 2 * HH + t] * f2;

    atomicAdd(&s[(size_t)r * HH + t], ds);
    float u0 = uvec[e * 3 + 0], u1 = uvec[e * 3 + 1], u2 = uvec[e * 3 + 2];
    size_t vc = (size_t)c * 3 * HH, vr = (size_t)r * 3 * HH;
    atomicAdd(&v[vr + t],          v_old[vc + t] * x1          + u0 * x2);
    atomicAdd(&v[vr + HH + t],     v_old[vc + HH + t] * x1     + u1 * x2);
    atomicAdd(&v[vr + 2 * HH + t], v_old[vc + 2 * HH + t] * x1 + u2 * x2);
}

// ---------------------------------------------------------------------------
// vn = ||Vv||_d ; cat = [s | vn]
// ---------------------------------------------------------------------------
__global__ void k_vn_cat(const float* __restrict__ s, const float* __restrict__ Vv,
                         float* __restrict__ cat)
{
    int i = blockIdx.x, t = threadIdx.x;
    size_t b = (size_t)i * 3 * HH;
    float a0 = Vv[b + t], a1 = Vv[b + HH + t], a2 = Vv[b + 2 * HH + t];
    cat[(size_t)i * 2 * HH + t]      = s[(size_t)i * HH + t];
    cat[(size_t)i * 2 * HH + HH + t] = sqrtf(a0 * a0 + a1 * a1 + a2 * a2 + 1e-8f);
}

// s += a_ss + a_sv * <Uv,Vv>_d ; v += a_vv * Uv
__global__ void k_update(float* __restrict__ s, float* __restrict__ v,
                         const float* __restrict__ Uv, const float* __restrict__ Vv,
                         const float* __restrict__ a)
{
    int i = blockIdx.x, t = threadIdx.x;
    size_t b = (size_t)i * 3 * HH;
    float dot = 0.f;
#pragma unroll
    for (int d = 0; d < 3; ++d) dot = fmaf(Uv[b + d * HH + t], Vv[b + d * HH + t], dot);
    size_t ab = (size_t)i * 3 * HH;
    float ass = a[ab + t], asv = a[ab + HH + t], avv = a[ab + 2 * HH + t];
    s[(size_t)i * HH + t] += ass + asv * dot;
#pragma unroll
    for (int d = 0; d < 3; ++d) v[b + d * HH + t] += avv * Uv[b + d * HH + t];
}

// ---------------------------------------------------------------------------
// Residue scatter-mean
// ---------------------------------------------------------------------------
__global__ void k_res_accum(const float* __restrict__ s, const int* __restrict__ ridx,
                            float* __restrict__ sum, float* __restrict__ cnt)
{
    int i = blockIdx.x, t = threadIdx.x;
    int r = ridx[i];
    atomicAdd(&sum[(size_t)r * HH + t], s[(size_t)i * HH + t]);
    if (t == 0) atomicAdd(&cnt[r], 1.f);
}

__global__ void k_res_div(float* __restrict__ sum, const float* __restrict__ cnt)
{
    int r = blockIdx.x, t = threadIdx.x;
    sum[(size_t)r * HH + t] /= fmaxf(cnt[r], 1.f);
}

// pos_pred = fake_pos + v2 . v_proj
__global__ void k_finalize(const float* __restrict__ v2, const float* __restrict__ vproj,
                           const float* __restrict__ fpos, float* __restrict__ out, int n3)
{
    int i = blockIdx.x * blockDim.x + threadIdx.x;
    if (i >= n3) return;
    int node = i / 3, d = i % 3;
    const float* row = v2 + ((size_t)node * 3 + d) * HH;
    float acc = 0.f;
    for (int h = 0; h < HH; ++h) acc = fmaf(row[h], vproj[h], acc);
    out[i] = fpos[i] + acc;
}

// ---------------------------------------------------------------------------
// Host side
// ---------------------------------------------------------------------------
struct Dense { const float* w; const float* b; };
struct Layer { Dense phi1, phi2, filt; const float* U; const float* V; Dense a1, a2; };
struct PackedLayer { const __bf16 *phi1, *phi2, *U, *V, *a1, *a2; };

static void parse_layers(void* const* d_in, int& i, Layer* L, int n)
{
    for (int l = 0; l < n; ++l) {
        L[l].phi1 = { (const float*)d_in[i], (const float*)d_in[i + 1] }; i += 2;
        L[l].phi2 = { (const float*)d_in[i], (const float*)d_in[i + 1] }; i += 2;
        L[l].filt = { (const float*)d_in[i], (const float*)d_in[i + 1] }; i += 2;
        L[l].U = (const float*)d_in[i++];
        L[l].V = (const float*)d_in[i++];
        L[l].a1 = { (const float*)d_in[i], (const float*)d_in[i + 1] }; i += 2;
        L[l].a2 = { (const float*)d_in[i], (const float*)d_in[i + 1] }; i += 2;
    }
}

static void painn_layer(const Layer& lp, const PackedLayer& pk,
                        const float* eattr, const float* uvec,
                        const int* ei, float* s, float* v, float* v_old, float* phi,
                        float* h1, float* cat, float* a, float* Uv, float* Vv,
                        hipStream_t stream)
{
    launch_gemm<HH, 4, 1>(s, pk.phi1, lp.phi1.b, h1, NN, HH, stream);
    launch_gemm<HH, 6, 0>(h1, pk.phi2, lp.phi2.b, phi, NN, 3 * HH, stream);
    hipMemcpyAsync(v_old, v, (size_t)NN * 3 * HH * sizeof(float), hipMemcpyDeviceToDevice, stream);
    k_edge_msg<<<EE, HH, 0, stream>>>(ei, eattr, lp.filt.w, lp.filt.b, phi, uvec, v_old, s, v, EE);
    launch_gemm<HH, 4, 0>(v, pk.U, nullptr, Uv, 3 * NN, HH, stream);
    launch_gemm<HH, 4, 0>(v, pk.V, nullptr, Vv, 3 * NN, HH, stream);
    k_vn_cat<<<NN, HH, 0, stream>>>(s, Vv, cat);
    launch_gemm<2 * HH, 4, 1>(cat, pk.a1, lp.a1.b, h1, NN, HH, stream);
    launch_gemm<HH, 6, 0>(h1, pk.a2, lp.a2.b, a, NN, 3 * HH, stream);
    k_update<<<NN, HH, 0, stream>>>(s, v, Uv, Vv, a);
}

extern "C" void kernel_launch(void* const* d_in, const int* in_sizes, int n_in,
                              void* d_out, int out_size, void* d_ws, size_t ws_size,
                              hipStream_t stream)
{
    (void)in_sizes; (void)n_in; (void)out_size; (void)ws_size;

    int i = 0;
    const int*   z      = (const int*)d_in[i++];
    const float* vecf   = (const float*)d_in[i++];
    const int*   ei     = (const int*)d_in[i++];
    const float* eattr  = (const float*)d_in[i++];
    const float* pos    = (const float*)d_in[i++];
    const int*   ridx   = (const int*)d_in[i++];
    const float* pnoise = (const float*)d_in[i++];

    const float* enc_embed = (const float*)d_in[i++];
    Layer enc[6]; parse_layers(d_in, i, enc, 6);
    const float* dec_embed = (const float*)d_in[i++];
    Layer dec[4]; parse_layers(d_in, i, dec, 4);
    const float* v_proj = (const float*)d_in[i++];
    Dense le1 = { (const float*)d_in[i], (const float*)d_in[i + 1] }; i += 2;
    Dense le2 = { (const float*)d_in[i], (const float*)d_in[i + 1] }; i += 2;
    Dense dmu = { (const float*)d_in[i], (const float*)d_in[i + 1] }; i += 2;
    Dense dlv = { (const float*)d_in[i], (const float*)d_in[i + 1] }; i += 2;
    Dense ld1 = { (const float*)d_in[i], (const float*)d_in[i + 1] }; i += 2;
    Dense ld2 = { (const float*)d_in[i], (const float*)d_in[i + 1] }; i += 2;
    Dense ld3 = { (const float*)d_in[i], (const float*)d_in[i + 1] }; i += 2;

    // Workspace carve-out (256B aligned slabs)
    char* base = (char*)d_ws;
    size_t off = 0;
    auto slab_b = [&](size_t bytes) -> void* {
        void* p = (void*)(base + off);
        off += (bytes + 255) & ~(size_t)255;
        return p;
    };
    auto slab = [&](size_t nfloats) -> float* {
        return (float*)slab_b(nfloats * sizeof(float));
    };

    float* s      = slab((size_t)NN * HH);
    float* v      = slab((size_t)NN * 3 * HH);
    float* v_old  = slab((size_t)NN * 3 * HH);
    float* phi    = slab((size_t)NN * 3 * HH);
    float* h1     = slab((size_t)NN * HH);
    float* cat    = slab((size_t)NN * 2 * HH);
    float* abuf   = slab((size_t)NN * 3 * HH);
    float* Uv     = slab((size_t)NN * 3 * HH);
    float* Vv     = slab((size_t)NN * 3 * HH);
    float* uvec_e = slab((size_t)EE * 3);
    float* uvec_d = slab((size_t)EE * 3);
    float* fea    = slab((size_t)EE * EDD);
    float* fpos   = slab((size_t)NN * 3);
    float* rsum   = slab((size_t)RR * HH);
    float* rcnt   = slab((size_t)RR);
    float* hR1    = slab((size_t)RR * HH);
    float* hR2    = slab((size_t)RR * HH);
    float* rf     = slab((size_t)RR * HH);

    // ---------------- Weight packing (bf16, fragment order) ----------------
    auto pack = [&](const float* W, int K, int M) -> const __bf16* {
        __bf16* p = (__bf16*)slab_b((size_t)K * M * sizeof(__bf16));
        k_pack_w<<<(K / 32) * M, 32, 0, stream>>>(W, p, M);
        return p;
    };
    PackedLayer penc[6], pdec[4];
    for (int l = 0; l < 6; ++l) {
        penc[l].phi1 = pack(enc[l].phi1.w, HH, HH);
        penc[l].phi2 = pack(enc[l].phi2.w, HH, 3 * HH);
        penc[l].U    = pack(enc[l].U, HH, HH);
        penc[l].V    = pack(enc[l].V, HH, HH);
        penc[l].a1   = pack(enc[l].a1.w, 2 * HH, HH);
        penc[l].a2   = pack(enc[l].a2.w, HH, 3 * HH);
    }
    for (int l = 0; l < 4; ++l) {
        pdec[l].phi1 = pack(dec[l].phi1.w, HH, HH);
        pdec[l].phi2 = pack(dec[l].phi2.w, HH, 3 * HH);
        pdec[l].U    = pack(dec[l].U, HH, HH);
        pdec[l].V    = pack(dec[l].V, HH, HH);
        pdec[l].a1   = pack(dec[l].a1.w, 2 * HH, HH);
        pdec[l].a2   = pack(dec[l].a2.w, HH, 3 * HH);
    }
    const __bf16* ple1 = pack(le1.w, HH, HH);
    const __bf16* ple2 = pack(le2.w, HH, HH);
    const __bf16* pmu  = pack(dmu.w, HH, LL);
    const __bf16* plv  = pack(dlv.w, HH, LL);
    const __bf16* pld1 = pack(ld1.w, LL, HH);
    const __bf16* pld2 = pack(ld2.w, HH, HH);
    const __bf16* pld3 = pack(ld3.w, HH, HH);

    float* out      = (float*)d_out;
    float* out_mu   = out + (size_t)NN * 3;                    // 60000
    float* out_lv   = out_mu + (size_t)RR * LL;                // +64000

    // ---------------- Encoder ----------------
    k_init_enc<<<NN, HH, 0, stream>>>(z, enc_embed, vecf, s, v);
    k_edge_geom<<<(EE + 255) / 256, 256, 0, stream>>>(ei, pos, uvec_e, EE);
    for (int l = 0; l < 6; ++l)
        painn_layer(enc[l], penc[l], eattr, uvec_e, ei, s, v, v_old, phi, h1, cat,
                    abuf, Uv, Vv, stream);

    // ---------------- Residue pooling + VAE encoder ----------------
    hipMemsetAsync(rsum, 0, (size_t)RR * HH * sizeof(float), stream);
    hipMemsetAsync(rcnt, 0, (size_t)RR * sizeof(float), stream);
    k_res_accum<<<NN, HH, 0, stream>>>(s, ridx, rsum, rcnt);
    k_res_div<<<RR, HH, 0, stream>>>(rsum, rcnt);
    launch_gemm<HH, 4, 1>(rsum, ple1, le1.b, hR1, RR, HH, stream);
    launch_gemm<HH, 4, 1>(hR1, ple2, le2.b, hR2, RR, HH, stream);
    launch_gemm<HH, 2, 0>(hR2, pmu, dmu.b, out_mu, RR, LL, stream);
    launch_gemm<HH, 2, 0>(hR2, plv, dlv.b, out_lv, RR, LL, stream);

    // ---------------- Decoder prep ----------------
    k_fake_pos<<<(NN * 3 + 255) / 256, 256, 0, stream>>>(pos, pnoise, fpos, NN * 3);
    k_fake_edge<<<(EE + 255) / 256, 256, 0, stream>>>(ei, fpos, eattr, fea, uvec_d, EE);
    launch_gemm<LL, 4, 1>(out_mu, pld1, ld1.b, hR1, RR, HH, stream);
    launch_gemm<HH, 4, 1>(hR1, pld2, ld2.b, hR2, RR, HH, stream);
    launch_gemm<HH, 4, 0>(hR2, pld3, ld3.b, rf, RR, HH, stream);
    k_init_dec<<<NN, HH, 0, stream>>>(z, dec_embed, ridx, rf, s);
    hipMemsetAsync(v, 0, (size_t)NN * 3 * HH * sizeof(float), stream);

    // ---------------- Decoder ----------------
    for (int l = 0; l < 4; ++l)
        painn_layer(dec[l], pdec[l], fea, uvec_d, ei, s, v, v_old, phi, h1, cat,
                    abuf, Uv, Vv, stream);

    // ---------------- Output ----------------
    k_finalize<<<(NN * 3 + 255) / 256, 256, 0, stream>>>(v, v_proj, fpos, out, NN * 3);
}
